// AttentionBlock_10024453669608
// MI455X (gfx1250) — matrix operand order repaired and checked by reading, not compiled, not run on hardware
//
#include <hip/hip_runtime.h>

typedef __attribute__((ext_vector_type(16))) _Float16 v16h;
typedef __attribute__((ext_vector_type(8)))  _Float16 v8h;
typedef __attribute__((ext_vector_type(8)))  float    v8f;

#define WMMA_F16(a, b, c) __builtin_amdgcn_wmma_f32_16x16x32_f16( \
    false, (a), false, (b), (short)0, (c), false, false)

#define BATCH 8
#define CCH   256
#define HW    4096
#define NGRP  32

// ---------------------------------------------------------------------------
// Fragment loaders (contiguous k = 16*(L/16) + s order used by this file, not the ISA table order; wave32)
// A 16x32 f16 fragment: lane L holds row M=L%16; lanes 0-15: K=0..15;
// lanes 16-31: K=16..31.
__device__ __forceinline__ v16h load_a_frag(const _Float16* base, int stride, int lane) {
    const _Float16* p = base + (size_t)(lane & 15) * stride + ((lane & 16) ? 16 : 0);
    v8h lo = *(const v8h*)(p);
    v8h hi = *(const v8h*)(p + 8);
    v16h r;
#pragma unroll
    for (int i = 0; i < 8; ++i) { r[i] = lo[i]; r[i + 8] = hi[i]; }
    return r;
}

// B 32x16 f16 fragment from N-major/K-contiguous storage ("NT"):
// lane L holds column N=L%16; lanes 0-15: K=0..15, lanes 16-31: K=16..31.
__device__ __forceinline__ v16h load_b_frag(const _Float16* base, int stride, int lane) {
    const _Float16* p = base + (size_t)(lane & 15) * stride + ((lane & 16) ? 16 : 0);
    return *(const v16h*)p;
}

// ---------------------------------------------------------------------------
// K0: convert weights f32 -> f16 (keep (o, c) layout = WMMA B-operand layout)
__global__ void wconv_kernel(const float* __restrict__ qkvw, const float* __restrict__ outw,
                             _Float16* __restrict__ whq, _Float16* __restrict__ who) {
    int i = blockIdx.x * 256 + threadIdx.x;
    if (i < 768 * CCH) whq[i] = (_Float16)qkvw[i];
    if (i < CCH * CCH) who[i] = (_Float16)outw[i];
}

// ---------------------------------------------------------------------------
// K1: GroupNorm statistics. One block per (b, group); reduce 8*4096 floats.
__global__ __launch_bounds__(256) void gn_stats_kernel(const float* __restrict__ x,
                                                       float* __restrict__ mean,
                                                       float* __restrict__ rstd) {
    int g = blockIdx.x;              // 0..255 = b*32 + grp
    int b = g >> 5, grp = g & 31;
    const float* base = x + ((size_t)b * CCH + (size_t)grp * 8) * HW;
    float s = 0.f, s2 = 0.f;
    for (int i = threadIdx.x; i < 8 * HW; i += 256) {
        float v = base[i];
        s += v; s2 += v * v;
    }
    __shared__ float sh[256], sh2[256];
    sh[threadIdx.x] = s; sh2[threadIdx.x] = s2;
    __syncthreads();
    for (int off = 128; off > 0; off >>= 1) {
        if (threadIdx.x < off) {
            sh[threadIdx.x]  += sh[threadIdx.x + off];
            sh2[threadIdx.x] += sh2[threadIdx.x + off];
        }
        __syncthreads();
    }
    if (threadIdx.x == 0) {
        float m = sh[0] * (1.f / 32768.f);
        float var = sh2[0] * (1.f / 32768.f) - m * m;
        mean[g] = m;
        rstd[g] = rsqrtf(var + 1e-5f);
    }
}

// K2: apply GroupNorm, convert to f16, transpose to xnT(b, p, c).
// 64x64 tile transpose via LDS: coalesced f32 reads along p, coalesced f16
// writes along c (avoids 512B-stride 2-byte scatter = ~32x write amplification).
__global__ __launch_bounds__(256) void gn_apply_kernel(
    const float* __restrict__ x, const float* __restrict__ mean,
    const float* __restrict__ rstd, const float* __restrict__ gw,
    const float* __restrict__ gb, _Float16* __restrict__ xnT) {
    __shared__ _Float16 tile[64][65];          // [p][c], padded
    int b  = blockIdx.z;
    int c0 = blockIdx.y * 64;
    int p0 = blockIdx.x * 64;
    int tx = threadIdx.x & 63;                 // p offset (read phase)
    int ty = threadIdx.x >> 6;                 // c offset (read phase)
#pragma unroll
    for (int cc = 0; cc < 64; cc += 4) {
        int c = c0 + cc + ty;
        int g = b * NGRP + (c >> 3);
        float v = x[((size_t)b * CCH + c) * HW + p0 + tx];
        v = (v - mean[g]) * rstd[g] * gw[c] + gb[c];
        tile[tx][cc + ty] = (_Float16)v;
    }
    __syncthreads();
    int cx = threadIdx.x & 63;                 // c offset (write phase)
    int py = threadIdx.x >> 6;                 // p offset (write phase)
#pragma unroll
    for (int pp = 0; pp < 64; pp += 4) {
        xnT[((size_t)b * HW + p0 + pp + py) * CCH + c0 + cx] = tile[pp + py][cx];
    }
}

// ---------------------------------------------------------------------------
// K3: QKV GEMM per batch: C(4096 x 768) = xnT(4096 x 256) * W(768 x 256)^T
// 8 waves/block, each wave: 16(M) x 64(N) tile, K=256 in 8 steps of 32.
__global__ __launch_bounds__(256) void gemm_qkv_kernel(
    const _Float16* __restrict__ A, const _Float16* __restrict__ W,
    const float* __restrict__ bias,
    _Float16* __restrict__ qb, _Float16* __restrict__ kb, _Float16* __restrict__ vb) {
    int lane = threadIdx.x & 31, wave = threadIdx.x >> 5;
    int b  = blockIdx.z;
    int m0 = blockIdx.y * 128 + wave * 16;
    int n0 = blockIdx.x * 64;
    const _Float16* Arow = A + ((size_t)b * HW + m0) * CCH;
    v8f acc[4] = {};
#pragma unroll
    for (int kc = 0; kc < 8; ++kc) {
        v16h a = load_a_frag(Arow + kc * 32, CCH, lane);
#pragma unroll
        for (int nn = 0; nn < 4; ++nn) {
            v16h bf = load_b_frag(W + (size_t)(n0 + nn * 16) * CCH + kc * 32, CCH, lane);
            acc[nn] = WMMA_F16(a, bf, acc[nn]);
        }
    }
#pragma unroll
    for (int nn = 0; nn < 4; ++nn) {
        int o = n0 + nn * 16 + (lane & 15);
        float bv = bias[o];
        if (o >= 512) {
            // V transposed store: each lane owns 8 consecutive p -> one 16B store
            v8h hv;
#pragma unroll
            for (int r = 0; r < 8; ++r) hv[r] = (_Float16)(acc[nn][r] + bv);
            size_t off = ((size_t)b * CCH + (o - 512)) * HW + m0 + ((lane & 16) ? 8 : 0);
            *(v8h*)(vb + off) = hv;
        } else {
#pragma unroll
            for (int r = 0; r < 8; ++r) {
                int p = m0 + r + ((lane & 16) ? 8 : 0);
                _Float16 hv = (_Float16)(acc[nn][r] + bv);
                if (o < 256) qb[((size_t)b * HW + p) * CCH + o]         = hv;
                else         kb[((size_t)b * HW + p) * CCH + (o - 256)] = hv;
            }
        }
    }
}

// ---------------------------------------------------------------------------
// K4: flash attention. 4 waves/block; wave pair shares 16 q-rows (Q in LDS),
// the two waves split the 256 output channels (8 f32 C-tiles each).
__global__ __launch_bounds__(128) void flash_kernel(
    const _Float16* __restrict__ q, const _Float16* __restrict__ kT,
    const _Float16* __restrict__ vT, _Float16* __restrict__ o) {
    __shared__ __align__(32) _Float16 qs[2][16 * CCH];   // 16 KB
    __shared__ __align__(32) _Float16 ps[4][16 * 32];    // 4 KB
    int lane = threadIdx.x & 31;
    int wave = threadIdx.x >> 5;
    int rowgrp = wave >> 1;
    int chalf  = wave & 1;
    int b  = blockIdx.y;
    int i0 = blockIdx.x * 32 + rowgrp * 16;

    { // cooperative Q stage: pair loads 16x256 halves = 512 x v8h
        const v8h* src = (const v8h*)(q + ((size_t)b * HW + i0) * CCH);
        v8h* dst = (v8h*)qs[rowgrp];
        for (int t = lane + chalf * 32; t < 512; t += 64) dst[t] = src[t];
    }
    __syncthreads();

    v8f oacc[8] = {};
    float m[8], l[8];
#pragma unroll
    for (int r = 0; r < 8; ++r) { m[r] = -1e30f; l[r] = 0.f; }
    const float scale = 0.0625f;   // 1/sqrt(256)

    for (int j0 = 0; j0 < HW; j0 += 32) {
        const _Float16* krow = kT + ((size_t)b * HW + j0) * CCH;

        { // prefetch next j-tile's K (32 rows x 512B) and V (128 rows x 64B)
            int jn = (j0 + 32) & (HW - 1);
            const char* kn = (const char*)(kT + ((size_t)b * HW + jn) * CCH + (size_t)lane * CCH);
#pragma unroll
            for (int pf = 0; pf < 4; ++pf) __builtin_prefetch(kn + pf * 128, 0, 3);
            const char* vn = (const char*)(vT + ((size_t)b * CCH + chalf * 128 + lane) * HW + jn);
#pragma unroll
            for (int pf = 0; pf < 4; ++pf) __builtin_prefetch(vn + (size_t)pf * 32 * HW * 2, 0, 3);
        }

        v8f s0 = {}, s1 = {};
#pragma unroll
        for (int kc = 0; kc < 8; ++kc) {
            v16h a  = load_a_frag(qs[rowgrp] + kc * 32, CCH, lane);
            v16h b0 = load_b_frag(krow + kc * 32, CCH, lane);
            v16h b1 = load_b_frag(krow + (size_t)16 * CCH + kc * 32, CCH, lane);
            s0 = WMMA_F16(a, b0, s0);
            s1 = WMMA_F16(a, b1, s1);
        }
        float alpha[8];
#pragma unroll
        for (int r = 0; r < 8; ++r) {
            float mx = fmaxf(s0[r], s1[r]);
            mx = fmaxf(mx, __shfl_xor(mx, 1));
            mx = fmaxf(mx, __shfl_xor(mx, 2));
            mx = fmaxf(mx, __shfl_xor(mx, 4));
            mx = fmaxf(mx, __shfl_xor(mx, 8));
            mx *= scale;
            float mn = fmaxf(m[r], mx);
            alpha[r] = __expf(m[r] - mn);
            m[r] = mn;
            float p0 = __expf(s0[r] * scale - mn);
            float p1 = __expf(s1[r] * scale - mn);
            float rs = p0 + p1;
            rs += __shfl_xor(rs, 1);
            rs += __shfl_xor(rs, 2);
            rs += __shfl_xor(rs, 4);
            rs += __shfl_xor(rs, 8);
            l[r] = l[r] * alpha[r] + rs;
            int row = r + ((lane & 16) ? 8 : 0);
            ps[wave][row * 32 + (lane & 15)]      = (_Float16)p0;
            ps[wave][row * 32 + (lane & 15) + 16] = (_Float16)p1;
        }
#pragma unroll
        for (int cc = 0; cc < 8; ++cc)
#pragma unroll
            for (int r = 0; r < 8; ++r) oacc[cc][r] *= alpha[r];

        __builtin_amdgcn_wave_barrier();
        asm volatile("s_wait_dscnt 0x0" ::: "memory");
        v16h pf = load_a_frag(ps[wave], 32, lane);
        const _Float16* vbase = vT + ((size_t)b * CCH + chalf * 128) * HW + j0;
#pragma unroll
        for (int cc = 0; cc < 8; ++cc) {
            v16h bf = load_b_frag(vbase + (size_t)cc * 16 * HW, HW, lane);
            oacc[cc] = WMMA_F16(pf, bf, oacc[cc]);
        }
        __builtin_amdgcn_wave_barrier();
    }

    float inv[8];
#pragma unroll
    for (int r = 0; r < 8; ++r) inv[r] = 1.f / l[r];
#pragma unroll
    for (int cc = 0; cc < 8; ++cc) {
        int c = chalf * 128 + cc * 16 + (lane & 15);
#pragma unroll
        for (int r = 0; r < 8; ++r) {
            int i = i0 + r + ((lane & 16) ? 8 : 0);
            o[((size_t)b * HW + i) * CCH + c] = (_Float16)(oacc[cc][r] * inv[r]);
        }
    }
}

// ---------------------------------------------------------------------------
// K5: out projection + bias + residual. Each lane owns 8 consecutive p for a
// fixed output channel -> 32B vector load of residual x and 32B vector store.
__global__ __launch_bounds__(256) void gemm_out_kernel(
    const _Float16* __restrict__ A, const _Float16* __restrict__ W,
    const float* __restrict__ bias, const float* __restrict__ x,
    float* __restrict__ out) {
    int lane = threadIdx.x & 31, wave = threadIdx.x >> 5;
    int b  = blockIdx.z;
    int m0 = blockIdx.y * 128 + wave * 16;
    int n0 = blockIdx.x * 64;
    const _Float16* Arow = A + ((size_t)b * HW + m0) * CCH;
    v8f acc[4] = {};
#pragma unroll
    for (int kc = 0; kc < 8; ++kc) {
        v16h a = load_a_frag(Arow + kc * 32, CCH, lane);
#pragma unroll
        for (int nn = 0; nn < 4; ++nn) {
            v16h bf = load_b_frag(W + (size_t)(n0 + nn * 16) * CCH + kc * 32, CCH, lane);
            acc[nn] = WMMA_F16(a, bf, acc[nn]);
        }
    }
#pragma unroll
    for (int nn = 0; nn < 4; ++nn) {
        int oc = n0 + nn * 16 + (lane & 15);
        float bv = bias[oc];
        size_t idx = ((size_t)b * CCH + oc) * HW + m0 + ((lane & 16) ? 8 : 0);
        v8f xv = *(const v8f*)(x + idx);
        v8f res;
#pragma unroll
        for (int r = 0; r < 8; ++r) res[r] = acc[nn][r] + bv + xv[r];
        *(v8f*)(out + idx) = res;
    }
}

// ---------------------------------------------------------------------------
extern "C" void kernel_launch(void* const* d_in, const int* in_sizes, int n_in,
                              void* d_out, int out_size, void* d_ws, size_t ws_size,
                              hipStream_t stream) {
    const float* x    = (const float*)d_in[0];
    const float* gw   = (const float*)d_in[1];
    const float* gb   = (const float*)d_in[2];
    const float* qkvw = (const float*)d_in[3];
    const float* qkvb = (const float*)d_in[4];
    const float* outw = (const float*)d_in[5];
    const float* outb = (const float*)d_in[6];
    float* out = (float*)d_out;

    char* ws = (char*)d_ws;
    const size_t SZ_BPC = (size_t)BATCH * HW * CCH * sizeof(_Float16);  // 16 MB each
    float*    mean = (float*)(ws);                                  // 1 KB
    float*    rstd = (float*)(ws + 1024);                           // 1 KB
    _Float16* whq  = (_Float16*)(ws + 2048);                        // 768*256*2
    _Float16* who  = (_Float16*)(ws + 2048 + 768 * CCH * 2);        // 256*256*2
    char*     big  = ws + 2048 + 768 * CCH * 2 + CCH * CCH * 2;
    _Float16* xnT  = (_Float16*)(big);
    _Float16* qb   = (_Float16*)(big + 1 * SZ_BPC);
    _Float16* kb   = (_Float16*)(big + 2 * SZ_BPC);
    _Float16* vb   = (_Float16*)(big + 3 * SZ_BPC);
    _Float16* ao   = (_Float16*)(big + 4 * SZ_BPC);

    wconv_kernel<<<(768 * CCH + 255) / 256, 256, 0, stream>>>(qkvw, outw, whq, who);
    gn_stats_kernel<<<BATCH * NGRP, 256, 0, stream>>>(x, mean, rstd);
    gn_apply_kernel<<<dim3(HW / 64, CCH / 64, BATCH), 256, 0, stream>>>(x, mean, rstd, gw, gb, xnT);
    gemm_qkv_kernel<<<dim3(768 / 64, HW / 128, BATCH), 256, 0, stream>>>(xnT, whq, qkvb, qb, kb, vb);
    flash_kernel<<<dim3(HW / 32, BATCH), 128, 0, stream>>>(qb, kb, vb, ao);
    gemm_out_kernel<<<dim3(CCH / 64, HW / 128, BATCH), 256, 0, stream>>>(ao, who, outb, x, out);
}
